// CrossAttnPromptModel_2714419331464
// MI455X (gfx1250) — matrix operand order repaired and checked by reading, not compiled, not run on hardware
//
#include <hip/hip_runtime.h>
#include <hip/hip_bf16.h>
#include <math.h>

// ---------------------------------------------------------------------------
// CDNA5 / gfx1250: wave32, WMMA 16x16x32 bf16 (f32 accum).
// Dead code eliminated vs reference: ctx-attention (#2) and ctx-MLP never
// influence the returned x, so only x-path attention + x-MLP are computed.
// All GEMM B operands are N-major (weights transpose-converted to bf16 once;
// V is produced transposed by the V-projection epilogue), so LDS staging is
// pure b128 traffic with no scalar scatter.
// ---------------------------------------------------------------------------

typedef __attribute__((ext_vector_type(8)))  float   v8f;
typedef __attribute__((ext_vector_type(16))) __bf16  bf16x16;
typedef __attribute__((ext_vector_type(8)))  __bf16  bf16x8;

#define N_IMG 1369     // 37*37 image tokens
#define N_PRM 2304     // 48*48 prompt tokens
#define DIMC  1024
#define HIDC  512
#define KVC   2048
#define MLPC  4096
#define BATCH 4
#define NHEAD 4
#define HEADD 256

#define BM 64
#define BN 128
#define BK 32

enum { EPI_F32 = 0, EPI_BIAS_BF16 = 1, EPI_BIAS_GELU_BF16 = 2,
       EPI_BIAS_RES_F32 = 3, EPI_BF16 = 4, EPI_BIAS_BF16_T = 5 };

__device__ __forceinline__ float gelu_exact(float v) {
  return 0.5f * v * (1.0f + erff(v * 0.70710678118654752f));
}

// ---------------------------------------------------------------------------
// Tiled bf16 WMMA GEMM:  Out = A(MxK) * B^T  with B stored N-major (N x K),
// [+bias] [+res], batched over z = (zb, zh) for per-(batch,head) GEMMs.
// ---------------------------------------------------------------------------
template<int EPI>
__global__ __launch_bounds__(256)
void gemm_wmma(const __bf16* __restrict__ A, const __bf16* __restrict__ Bm,
               const float* __restrict__ bias, const float* __restrict__ Res,
               void* __restrict__ OutV,
               int Mdim, int Ndim, int Kdim,
               int lda, int ldb, int ldc, int ldres,
               long sAz, long sBz, long sCz, long sResz,
               int Zh, long sAh, long sBh, long sCh)
{
  __shared__ __bf16 sA[BM][BK + 8];   // row-major A tile
  __shared__ __bf16 sB[BN][BK + 8];   // B tile [n][k] -> contiguous fragments

  const int tid  = threadIdx.x;
  const int lane = tid & 31;
  const int wave = tid >> 5;          // 8 waves
  const int waveM = wave >> 2;        // 0..1  (32-row strips)
  const int waveN = wave & 3;         // 0..3  (32-col strips)
  const int r16 = lane & 15;
  const int hi  = lane >> 4;          // lane half (ISA K-interleave select)

  const int zb = blockIdx.z / Zh;
  const int zh = blockIdx.z % Zh;
  A  += (long)zb * sAz + (long)zh * sAh;
  Bm += (long)zb * sBz + (long)zh * sBh;
  const long coff = (long)zb * sCz + (long)zh * sCh;
  if (Res) Res += (long)zb * sResz;

  const int m0 = blockIdx.y * BM;
  const int n0 = blockIdx.x * BN;

  v8f acc[2][2] = {};
  const int nK = Kdim / BK;

  for (int kb = 0; kb < nK; ++kb) {
    const int k0 = kb * BK;
    __syncthreads();
    // ---- stage A tile: 64x32 bf16, one b128 load/store per thread ----
    {
      const int rr = tid >> 2;              // 0..63
      const int cc = (tid & 3) * 8;         // 0,8,16,24
      bf16x8 v = {};
      const int grow = m0 + rr;
      if (grow < Mdim)
        v = *(const bf16x8*)(A + (long)grow * lda + k0 + cc);
      *(bf16x8*)&sA[rr][cc] = v;
    }
    // ---- stage B tile: 128x32 bf16, N-major rows -> direct b128 copies ----
#pragma unroll
    for (int c = tid; c < (BN * BK) / 8; c += 256) {
      const int nn = c >> 2;                // 0..127
      const int kk = (c & 3) * 8;
      bf16x8 v = {};
      const int gn = n0 + nn;
      if (gn < Ndim)
        v = *(const bf16x8*)(Bm + (long)gn * ldb + k0 + kk);
      *(bf16x8*)&sB[nn][kk] = v;
    }
    // prefetch next K tile of A into caches (global_prefetch_b8)
    if (kb + 1 < nK) {
      const int rr = tid >> 2;
      const int grow = m0 + rr;
      if (grow < Mdim)
        __builtin_prefetch(A + (long)grow * lda + k0 + BK, 0, 1);
    }
    __syncthreads();

    // ---- fragments per ISA 16-bit layouts ----
    bf16x16 afr[2], bfr[2];
#pragma unroll
    for (int mt = 0; mt < 2; ++mt) {
      const int rowT = waveM * 32 + mt * 16 + r16;
      bf16x8 lo = *(const bf16x8*)&sA[rowT][hi * 8];        // K  0..7 / 8..15
      bf16x8 up = *(const bf16x8*)&sA[rowT][16 + hi * 8];   // K 16..23 / 24..31
#pragma unroll
      for (int j = 0; j < 8; ++j) { afr[mt][j] = lo[j]; afr[mt][8 + j] = up[j]; }
    }
#pragma unroll
    for (int nt = 0; nt < 2; ++nt) {
      const int colT = waveN * 32 + nt * 16 + r16;
      bf16x8 lo = *(const bf16x8*)&sB[colT][hi * 8];        // K  0..7 / 8..15
      bf16x8 up = *(const bf16x8*)&sB[colT][16 + hi * 8];   // K 16..23 / 24..31
#pragma unroll
      for (int j = 0; j < 8; ++j) { bfr[nt][j] = lo[j]; bfr[nt][8 + j] = up[j]; }
    }
#pragma unroll
    for (int mt = 0; mt < 2; ++mt)
#pragma unroll
      for (int nt = 0; nt < 2; ++nt)
        acc[mt][nt] = __builtin_amdgcn_wmma_f32_16x16x32_bf16(
            false, afr[mt], false, bfr[nt], (short)0, acc[mt][nt], false, false);
  }

  // ---- epilogue (C layout: VGPR i -> row hi*8+i, col = lane&15) ----
#pragma unroll
  for (int mt = 0; mt < 2; ++mt) {
#pragma unroll
    for (int nt = 0; nt < 2; ++nt) {
      const int col = n0 + waveN * 32 + nt * 16 + r16;
      if (col >= Ndim) continue;
      float bv = 0.0f;
      if (EPI == EPI_BIAS_BF16 || EPI == EPI_BIAS_GELU_BF16 ||
          EPI == EPI_BIAS_RES_F32 || EPI == EPI_BIAS_BF16_T)
        bv = bias[col];
#pragma unroll
      for (int i = 0; i < 8; ++i) {
        const int row = m0 + waveM * 32 + mt * 16 + hi * 8 + i;
        if (row >= Mdim) continue;
        float v = acc[mt][nt][i] + bv;
        if (EPI == EPI_BIAS_BF16_T) {
          // transposed store: out[col][row] (used to produce V^T)
          ((__bf16*)OutV)[coff + (long)col * ldc + row] = (__bf16)v;
          continue;
        }
        const long idx = coff + (long)row * ldc + col;
        if (EPI == EPI_F32) {
          ((float*)OutV)[idx] = v;
        } else if (EPI == EPI_BIAS_BF16 || EPI == EPI_BF16) {
          ((__bf16*)OutV)[idx] = (__bf16)v;
        } else if (EPI == EPI_BIAS_GELU_BF16) {
          ((__bf16*)OutV)[idx] = (__bf16)gelu_exact(v);
        } else { // EPI_BIAS_RES_F32
          v += Res[(long)row * ldres + col];
          ((float*)OutV)[idx] = v;
        }
      }
    }
  }
}

// ---------------------------------------------------------------------------
// Elementwise / reduction kernels
// ---------------------------------------------------------------------------

// transpose-convert: in (K x N, f32) -> out (N x K, bf16); K,N multiples of 32
__global__ __launch_bounds__(256)
void f2bf_t_kernel(const float* __restrict__ in, __bf16* __restrict__ out,
                   int K, int N) {
  __shared__ float tile[32][33];
  const int k0 = blockIdx.y * 32;
  const int n0 = blockIdx.x * 32;
#pragma unroll
  for (int j = 0; j < 32; j += 8)
    tile[threadIdx.y + j][threadIdx.x] =
        in[(long)(k0 + threadIdx.y + j) * N + n0 + threadIdx.x];
  __syncthreads();
#pragma unroll
  for (int j = 0; j < 32; j += 8)
    out[(long)(n0 + threadIdx.y + j) * K + k0 + threadIdx.x] =
        (__bf16)tile[threadIdx.x][threadIdx.y + j];
}

// grid PE: out[p, j] = sin(2pi*c_j), out[p, 512+j] = cos(2pi*c_j)
__global__ void pe_kernel(const float* __restrict__ gauss, float* __restrict__ out,
                          int h, int w) {
  long idx = (long)blockIdx.x * 256 + threadIdx.x;
  long total = (long)h * w * HIDC;
  if (idx >= total) return;
  int  j = (int)(idx % HIDC);
  long p = idx / HIDC;
  int cw = (int)(p % w), rh = (int)(p / w);
  float xv = 2.0f * ((cw + 0.5f) / (float)w) - 1.0f;
  float yv = 2.0f * ((rh + 0.5f) / (float)h) - 1.0f;
  float t = 6.2831853071795864769f * (xv * gauss[j] + yv * gauss[HIDC + j]);
  out[p * DIMC + j]        = sinf(t);
  out[p * DIMC + HIDC + j] = cosf(t);
}

// h1[bm, j] = gelu(depth[bm] * w1[j] + b1[j])  -> bf16
__global__ void h1_kernel(const float* __restrict__ depth, const float* __restrict__ w1,
                          const float* __restrict__ b1, __bf16* __restrict__ out, long total) {
  long idx = (long)blockIdx.x * 256 + threadIdx.x;
  if (idx >= total) return;
  int  j  = (int)(idx % HIDC);
  long bm = idx / HIDC;
  out[idx] = (__bf16)gelu_exact(depth[bm] * w1[j] + b1[j]);
}

// out[b*S+i] = a[b*S+i] + pe[i % S]
__global__ void add_bcast_kernel(const float* __restrict__ a, const float* __restrict__ pe,
                                 float* __restrict__ out, long rowsz, long total) {
  long i = (long)blockIdx.x * 256 + threadIdx.x;
  if (i < total) out[i] = a[i] + pe[i % rowsz];
}

// LayerNorm over D=1024, one block per row, bf16 output for GEMM feeds
__global__ __launch_bounds__(256)
void ln_bf16_kernel(const float* __restrict__ X, const float* __restrict__ g,
                    const float* __restrict__ b, __bf16* __restrict__ Out) {
  __shared__ float red[256];
  const long row = blockIdx.x;
  const float* x = X + row * DIMC;
  float s = 0.0f;
  for (int i = threadIdx.x; i < DIMC; i += 256) s += x[i];
  red[threadIdx.x] = s; __syncthreads();
  for (int o = 128; o > 0; o >>= 1) { if (threadIdx.x < o) red[threadIdx.x] += red[threadIdx.x + o]; __syncthreads(); }
  const float mean = red[0] / DIMC;
  __syncthreads();
  float vs = 0.0f;
  for (int i = threadIdx.x; i < DIMC; i += 256) { float d = x[i] - mean; vs += d * d; }
  red[threadIdx.x] = vs; __syncthreads();
  for (int o = 128; o > 0; o >>= 1) { if (threadIdx.x < o) red[threadIdx.x] += red[threadIdx.x + o]; __syncthreads(); }
  const float inv = rsqrtf(red[0] / DIMC + 1e-5f);
  for (int i = threadIdx.x; i < DIMC; i += 256)
    Out[row * DIMC + i] = (__bf16)((x[i] - mean) * inv * g[i] + b[i]);
}

// masked softmax: grid(Nq, B*H); logits f32 (z,Nq,Nk) -> probs bf16
__global__ __launch_bounds__(256)
void softmax_kernel(const float* __restrict__ logits, const float* __restrict__ mask,
                    __bf16* __restrict__ probs, int Nq, int Nk, int Hh,
                    float scale, int useMask) {
  __shared__ float red[256];
  const int q = blockIdx.x;
  const int z = blockIdx.y;
  const int b = z / Hh;
  const long base = ((long)z * Nq + q) * Nk;
  const float* row = logits + base;
  const float* mrow = mask + (long)b * Nk;
  float mx = -3.4e38f;
  for (int i = threadIdx.x; i < Nk; i += 256) {
    float v = row[i] * scale;
    if (useMask && !(mrow[i] > 0.0f)) v = -1.0e9f;
    mx = fmaxf(mx, v);
  }
  red[threadIdx.x] = mx; __syncthreads();
  for (int o = 128; o > 0; o >>= 1) { if (threadIdx.x < o) red[threadIdx.x] = fmaxf(red[threadIdx.x], red[threadIdx.x + o]); __syncthreads(); }
  mx = red[0]; __syncthreads();
  float s = 0.0f;
  for (int i = threadIdx.x; i < Nk; i += 256) {
    float v = row[i] * scale;
    if (useMask && !(mrow[i] > 0.0f)) v = -1.0e9f;
    s += expf(v - mx);
  }
  red[threadIdx.x] = s; __syncthreads();
  for (int o = 128; o > 0; o >>= 1) { if (threadIdx.x < o) red[threadIdx.x] += red[threadIdx.x + o]; __syncthreads(); }
  const float inv = 1.0f / red[0];
  for (int i = threadIdx.x; i < Nk; i += 256) {
    float v = row[i] * scale;
    if (useMask && !(mrow[i] > 0.0f)) v = -1.0e9f;
    probs[base + i] = (__bf16)(expf(v - mx) * inv);
  }
}

// ---------------------------------------------------------------------------
// Orchestration
// ---------------------------------------------------------------------------
extern "C" void kernel_launch(void* const* d_in, const int* in_sizes, int n_in,
                              void* d_out, int out_size, void* d_ws, size_t ws_size,
                              hipStream_t stream) {
  (void)in_sizes; (void)n_in; (void)out_size; (void)ws_size;
  const float* emb    = (const float*)d_in[0];
  const float* pdepth = (const float*)d_in[1];
  const float* pmask  = (const float*)d_in[2];
  const float* gauss  = (const float*)d_in[5];
  const float* d2f_w1 = (const float*)d_in[6];
  const float* d2f_b1 = (const float*)d_in[7];
  const float* d2f_w2 = (const float*)d_in[8];
  const float* d2f_b2 = (const float*)d_in[9];
  const float* n1x_g  = (const float*)d_in[10];
  const float* n1x_b  = (const float*)d_in[11];
  const float* n1c_g  = (const float*)d_in[14];
  const float* n1c_b  = (const float*)d_in[15];
  const float* n2x_g  = (const float*)d_in[16];
  const float* n2x_b  = (const float*)d_in[17];
  const float* ax_wq  = (const float*)d_in[20];
  const float* ax_bq  = (const float*)d_in[21];
  const float* ax_wkv = (const float*)d_in[22];
  const float* ax_bkv = (const float*)d_in[23];
  const float* ax_wo  = (const float*)d_in[24];
  const float* ax_bo  = (const float*)d_in[25];
  const float* mx_w1  = (const float*)d_in[32];
  const float* mx_b1  = (const float*)d_in[33];
  const float* mx_w2  = (const float*)d_in[34];
  const float* mx_b2  = (const float*)d_in[35];

  // workspace carve-out
  char* ws = (char*)d_ws;
  size_t off = 0;
  auto take = [&](size_t bytes) { char* p = ws + off; off = (off + bytes + 255) & ~(size_t)255; return p; };

  __bf16* wq_t    = (__bf16*)take((size_t)DIMC * DIMC * 2);   // [out][in]
  __bf16* wkv_t   = (__bf16*)take((size_t)KVC  * DIMC * 2);   // [out(2048)][in]
  __bf16* wo_t    = (__bf16*)take((size_t)DIMC * DIMC * 2);
  __bf16* w1m_t   = (__bf16*)take((size_t)MLPC * DIMC * 2);   // [4096][1024]
  __bf16* w2m_t   = (__bf16*)take((size_t)DIMC * MLPC * 2);   // [1024][4096]
  __bf16* dw2_t   = (__bf16*)take((size_t)DIMC * HIDC * 2);   // [1024][512]
  float*  img_pe  = (float*) take((size_t)N_IMG * DIMC * 4);
  float*  prm_pe  = (float*) take((size_t)N_PRM * DIMC * 4);
  __bf16* h1_bf   = (__bf16*)take((size_t)BATCH * N_PRM * HIDC * 2);
  float*  ctx0    = (float*) take((size_t)BATCH * N_PRM * DIMC * 4);
  __bf16* cn_bf   = (__bf16*)take((size_t)BATCH * N_PRM * DIMC * 2);
  float*  ximg    = (float*) take((size_t)BATCH * N_IMG * DIMC * 4);
  __bf16* xn_bf   = (__bf16*)take((size_t)BATCH * N_IMG * DIMC * 2);
  __bf16* q_bf    = (__bf16*)take((size_t)BATCH * N_IMG * DIMC * 2);
  __bf16* k_bf    = (__bf16*)take((size_t)BATCH * N_PRM * DIMC * 2);  // [b][tok][1024]
  __bf16* vt_bf   = (__bf16*)take((size_t)BATCH * DIMC * N_PRM * 2);  // [b][d][tok]
  float*  logits  = (float*) take((size_t)BATCH * NHEAD * N_IMG * N_PRM * 4);
  __bf16* probs   = (__bf16*)take((size_t)BATCH * NHEAD * N_IMG * N_PRM * 2);
  __bf16* o_bf    = (__bf16*)take((size_t)BATCH * N_IMG * DIMC * 2);
  float*  x2      = (float*) take((size_t)BATCH * N_IMG * DIMC * 4);
  __bf16* xn2_bf  = (__bf16*)take((size_t)BATCH * N_IMG * DIMC * 2);
  __bf16* hmlp_bf = (__bf16*)take((size_t)BATCH * N_IMG * MLPC * 2);

  auto blocks = [](long n) { return (unsigned)((n + 255) / 256); };
  const dim3 tb(32, 8);

  // 1) weights -> bf16, transposed to N-major [out][in]
  f2bf_t_kernel<<<dim3(DIMC/32, DIMC/32), tb, 0, stream>>>(ax_wq,  wq_t,  DIMC, DIMC);
  f2bf_t_kernel<<<dim3(KVC /32, DIMC/32), tb, 0, stream>>>(ax_wkv, wkv_t, DIMC, KVC);
  f2bf_t_kernel<<<dim3(DIMC/32, DIMC/32), tb, 0, stream>>>(ax_wo,  wo_t,  DIMC, DIMC);
  f2bf_t_kernel<<<dim3(MLPC/32, DIMC/32), tb, 0, stream>>>(mx_w1,  w1m_t, DIMC, MLPC);
  f2bf_t_kernel<<<dim3(DIMC/32, MLPC/32), tb, 0, stream>>>(mx_w2,  w2m_t, MLPC, DIMC);
  f2bf_t_kernel<<<dim3(DIMC/32, HIDC/32), tb, 0, stream>>>(d2f_w2, dw2_t, HIDC, DIMC);

  // 2) positional encodings
  pe_kernel<<<blocks((long)N_IMG*HIDC), 256, 0, stream>>>(gauss, img_pe, 37, 37);
  pe_kernel<<<blocks((long)N_PRM*HIDC), 256, 0, stream>>>(gauss, prm_pe, 48, 48);

  // 3) depth -> hidden (gelu), then ctx0 = h1 @ d2f_w2 + b2 + prompt_pe
  h1_kernel<<<blocks((long)BATCH*N_PRM*HIDC), 256, 0, stream>>>(
      pdepth, d2f_w1, d2f_b1, h1_bf, (long)BATCH*N_PRM*HIDC);
  gemm_wmma<EPI_BIAS_RES_F32><<<dim3(DIMC/BN, (N_PRM+BM-1)/BM, BATCH), 256, 0, stream>>>(
      h1_bf, dw2_t, d2f_b2, prm_pe, ctx0,
      N_PRM, DIMC, HIDC, HIDC, HIDC, DIMC, DIMC,
      (long)N_PRM*HIDC, 0, (long)N_PRM*DIMC, 0, 1, 0, 0, 0);

  // 4) x = image_embeddings + image_pe ; layernorms -> bf16 GEMM inputs
  add_bcast_kernel<<<blocks((long)BATCH*N_IMG*DIMC), 256, 0, stream>>>(
      emb, img_pe, ximg, (long)N_IMG*DIMC, (long)BATCH*N_IMG*DIMC);
  ln_bf16_kernel<<<BATCH*N_PRM, 256, 0, stream>>>(ctx0, n1c_g, n1c_b, cn_bf);
  ln_bf16_kernel<<<BATCH*N_IMG, 256, 0, stream>>>(ximg, n1x_g, n1x_b, xn_bf);

  // 5) projections: Q; K (normal layout); V (transposed-store epilogue)
  gemm_wmma<EPI_BIAS_BF16><<<dim3(DIMC/BN, (N_IMG+BM-1)/BM, BATCH), 256, 0, stream>>>(
      xn_bf, wq_t, ax_bq, nullptr, q_bf,
      N_IMG, DIMC, DIMC, DIMC, DIMC, DIMC, 0,
      (long)N_IMG*DIMC, 0, (long)N_IMG*DIMC, 0, 1, 0, 0, 0);
  gemm_wmma<EPI_BIAS_BF16><<<dim3(DIMC/BN, (N_PRM+BM-1)/BM, BATCH), 256, 0, stream>>>(
      cn_bf, wkv_t, ax_bkv, nullptr, k_bf,
      N_PRM, DIMC, DIMC, DIMC, DIMC, DIMC, 0,
      (long)N_PRM*DIMC, 0, (long)N_PRM*DIMC, 0, 1, 0, 0, 0);
  gemm_wmma<EPI_BIAS_BF16_T><<<dim3(DIMC/BN, (N_PRM+BM-1)/BM, BATCH), 256, 0, stream>>>(
      cn_bf, wkv_t + (size_t)DIMC*DIMC, ax_bkv + DIMC, nullptr, vt_bf,
      N_PRM, DIMC, DIMC, DIMC, DIMC, N_PRM, 0,
      (long)N_PRM*DIMC, 0, (long)DIMC*N_PRM, 0, 1, 0, 0, 0);

  // 6) logits = Q * K^T  per (batch, head); K rows are token-major (N-major)
  gemm_wmma<EPI_F32><<<dim3(N_PRM/BN, (N_IMG+BM-1)/BM, BATCH*NHEAD), 256, 0, stream>>>(
      q_bf, k_bf, nullptr, nullptr, logits,
      N_IMG, N_PRM, HEADD, DIMC, DIMC, N_PRM, 0,
      (long)N_IMG*DIMC, (long)N_PRM*DIMC, (long)NHEAD*N_IMG*N_PRM, 0,
      NHEAD, HEADD, HEADD, (long)N_IMG*N_PRM);

  // 7) masked softmax (scale = 1/sqrt(256))
  softmax_kernel<<<dim3(N_IMG, BATCH*NHEAD), 256, 0, stream>>>(
      logits, pmask, probs, N_IMG, N_PRM, NHEAD, 0.0625f, 1);

  // 8) O = probs * V : V^T rows are head-dims (N-major), K = tokens
  gemm_wmma<EPI_BF16><<<dim3(HEADD/BN, (N_IMG+BM-1)/BM, BATCH*NHEAD), 256, 0, stream>>>(
      probs, vt_bf, nullptr, nullptr, o_bf,
      N_IMG, HEADD, N_PRM, N_PRM, N_PRM, DIMC, 0,
      (long)NHEAD*N_IMG*N_PRM, (long)DIMC*N_PRM, (long)N_IMG*DIMC, 0,
      NHEAD, (long)N_IMG*N_PRM, (long)HEADD*N_PRM, HEADD);

  // 9) x2 = O @ wo + bo + x   (residual)
  gemm_wmma<EPI_BIAS_RES_F32><<<dim3(DIMC/BN, (N_IMG+BM-1)/BM, BATCH), 256, 0, stream>>>(
      o_bf, wo_t, ax_bo, ximg, x2,
      N_IMG, DIMC, DIMC, DIMC, DIMC, DIMC, DIMC,
      (long)N_IMG*DIMC, 0, (long)N_IMG*DIMC, (long)N_IMG*DIMC, 1, 0, 0, 0);

  // 10) MLP: hidden = gelu(LN(x2) @ w1 + b1); out = hidden @ w2 + b2 + x2
  ln_bf16_kernel<<<BATCH*N_IMG, 256, 0, stream>>>(x2, n2x_g, n2x_b, xn2_bf);
  gemm_wmma<EPI_BIAS_GELU_BF16><<<dim3(MLPC/BN, (N_IMG+BM-1)/BM, BATCH), 256, 0, stream>>>(
      xn2_bf, w1m_t, mx_b1, nullptr, hmlp_bf,
      N_IMG, MLPC, DIMC, DIMC, DIMC, MLPC, 0,
      (long)N_IMG*DIMC, 0, (long)N_IMG*MLPC, 0, 1, 0, 0, 0);
  gemm_wmma<EPI_BIAS_RES_F32><<<dim3(DIMC/BN, (N_IMG+BM-1)/BM, BATCH), 256, 0, stream>>>(
      hmlp_bf, w2m_t, mx_b2, x2, (float*)d_out,
      N_IMG, DIMC, MLPC, MLPC, MLPC, DIMC, DIMC,
      (long)N_IMG*MLPC, 0, (long)N_IMG*DIMC, (long)N_IMG*DIMC, 1, 0, 0, 0);
}
